// RND_MPNNet_14834817040730
// MI455X (gfx1250) — compile-verified, matrix operand order checked by reading, not run on hardware
//
#include <hip/hip_runtime.h>
#include <math.h>

#define DIM 64
#define NUM_FEAT 14
#define EDGE_FEAT 4
#define OUT_DIM 16
#define NUM_GRAPHS 512
#define NEG_SLOPE 0.01f

typedef _Float16 v16h __attribute__((ext_vector_type(16)));
typedef float    v8f  __attribute__((ext_vector_type(8)));
typedef _Float16 h4   __attribute__((ext_vector_type(4)));

// ---- CDNA5 WMMA helpers (v_wmma_f32_16x16x32_f16) ---------------------------
// A (16x32 f16): lane L holds row M=L&15; elems e map to K = kidx(L,e)
// (elems 0..7 -> K {0..7 | 8..15}, elems 8..15 -> K {16..23 | 24..31} by lane half)
// B (32x16 f16): mirrored layout, lane L holds col N=L&15, same K element map.
// C/D (16x16 f32): elem r -> (M = r + 8*(L>=16), N = L&15).
__device__ __forceinline__ int kidx(int lane, int e) {
  return (e & 7) + ((e >> 3) << 4) + ((lane >> 4) << 3);
}
__device__ __forceinline__ v8f wmma32(v16h a, v16h b, v8f c) {
  return __builtin_amdgcn_wmma_f32_16x16x32_f16(false, a, false, b, (short)0, c,
                                                false, false);
}
__device__ __forceinline__ float leaky(float v) { return v >= 0.f ? v : NEG_SLOPE * v; }
__device__ __forceinline__ float sigm(float v) { return 1.f / (1.f + __expf(-v)); }

__device__ __forceinline__ void atomicMaxF(float* addr, float val) {
  int* ai = (int*)addr;
  int old = __float_as_int(*addr);
  while (__int_as_float(old) < val) {
    int assumed = old;
    old = atomicCAS(ai, assumed, __float_as_int(val));
    if (old == assumed) break;
  }
}

// ---- pack f32 weight matrix into WMMA-B f16 tile layout ---------------------
// pk[((chunk*nTiles + tile)*32 + lane)*16 + e] = W[k*sk + j*sj],
//   k = chunk*32 + kidx(lane,e), j = tile*16 + (lane&15)
__global__ void pack_b_kernel(const float* __restrict__ W, _Float16* __restrict__ pk,
                              int nTiles, int sk, int sj) {
  int idx = blockIdx.x * blockDim.x + threadIdx.x;
  int total = 2 * nTiles * 32 * 16;
  if (idx >= total) return;
  int e = idx & 15;
  int lane = (idx >> 4) & 31;
  int ct = idx >> 9;
  int t = ct % nTiles, c = ct / nTiles;
  int k = c * 32 + kidx(lane, e);
  int j = t * 16 + (lane & 15);
  pk[idx] = (_Float16)W[k * sk + j * sj];
}

// ---- edge network layer 1: hidden[E,64] = leaky(edge_attr @ W1^T + b1) ------
__global__ void edge_hidden_kernel(const float* __restrict__ ea,
                                   const float* __restrict__ w1,
                                   const float* __restrict__ b1,
                                   float* __restrict__ hid, int E) {
  int idx = blockIdx.x * blockDim.x + threadIdx.x;
  if (idx >= E * DIM) return;
  int e = idx >> 6, j = idx & 63;
  float acc = b1[j];
#pragma unroll
  for (int k = 0; k < EDGE_FEAT; k++) acc += ea[e * EDGE_FEAT + k] * w1[j * EDGE_FEAT + k];
  hid[idx] = leaky(acc);
}

// ---- ew[E,4096] (f16) = hidden[E,64] @ W2^T[64,4096] + b2 : WMMA GEMM -------
__global__ void __launch_bounds__(256)
ew_gemm_kernel(const float* __restrict__ hid, const _Float16* __restrict__ w2pk,
               const float* __restrict__ b2, _Float16* __restrict__ ew, int E) {
  __shared__ float sA[16 * 65];
  int e0 = blockIdx.x * 16;
  int tid = threadIdx.x;
  for (int i = tid; i < 16 * 64; i += 256) {
    int r = i >> 6, c = i & 63;
    sA[r * 65 + c] = (e0 + r < E) ? hid[(size_t)(e0 + r) * 64 + c] : 0.f;
  }
  __syncthreads();
  int lane = tid & 31, w = tid >> 5;
  int m = lane & 15;
  v16h a0, a1;
#pragma unroll
  for (int e = 0; e < 16; e++) {
    a0[e] = (_Float16)sA[m * 65 + kidx(lane, e)];
    a1[e] = (_Float16)sA[m * 65 + 32 + kidx(lane, e)];
  }
  int mb = (lane >> 4) << 3;
  const bool full = (e0 + 16 <= E);  // wave-uniform: straight-line stores
  for (int t = w; t < 256; t += 8) {
    v16h b0 = *(const v16h*)(w2pk + ((size_t)t * 32 + lane) * 16);
    v16h b1 = *(const v16h*)(w2pk + ((size_t)(256 + t) * 32 + lane) * 16);
    v8f c = {};
    c = wmma32(a0, b0, c);
    c = wmma32(a1, b1, c);
    int n = t * 16 + (lane & 15);
    float bias = b2[n];
    _Float16* wp = ew + (size_t)(e0 + mb) * 4096 + n;
    if (full) {
#pragma unroll
      for (int r = 0; r < 8; r++)
        __builtin_nontemporal_store((_Float16)(c[r] + bias), wp + (size_t)r * 4096);
    } else {
#pragma unroll
      for (int r = 0; r < 8; r++)
        if (e0 + mb + r < E)
          __builtin_nontemporal_store((_Float16)(c[r] + bias), wp + (size_t)r * 4096);
    }
  }
}

// ---- lin0: h[N,64] = leaky(x @ lin0_w^T + b) --------------------------------
__global__ void lin0_kernel(const float* __restrict__ x, const float* __restrict__ w,
                            const float* __restrict__ b, float* __restrict__ h, int N) {
  int idx = blockIdx.x * blockDim.x + threadIdx.x;
  if (idx >= N * DIM) return;
  int n = idx >> 6, j = idx & 63;
  float acc = b[j];
#pragma unroll
  for (int k = 0; k < NUM_FEAT; k++) acc += x[n * NUM_FEAT + k] * w[j * NUM_FEAT + k];
  h[idx] = leaky(acc);
}

__global__ void fill_kernel(float* __restrict__ p, float v, size_t n) {
  size_t i = (size_t)blockIdx.x * blockDim.x + threadIdx.x;
  if (i < n) p[i] = v;
}

__global__ void deg_kernel(const int* __restrict__ tgt, float* __restrict__ deg, int E) {
  int e = blockIdx.x * blockDim.x + threadIdx.x;
  if (e < E) atomicAdd(&deg[tgt[e]], 1.f);
}

// ---- per-edge matvec + scatter-add: agg[tgt] += h[src] @ ew_e ---------------
// one wave per edge; b64 nontemporal loads stream the 8KB per-edge matrix:
// lane halves take even/odd d rows, lane&15 picks 4 output cols; partial sums
// combined across the lane halves with a shfl_xor(16), then 64 atomics/edge.
__global__ void __launch_bounds__(256)
msg_scatter_kernel(const float* __restrict__ h, const _Float16* __restrict__ ew,
                   const int* __restrict__ src, const int* __restrict__ tgt,
                   float* __restrict__ agg, int E) {
  __shared__ float sh[8][64];
  int w = threadIdx.x >> 5, lane = threadIdx.x & 31;
  int e = blockIdx.x * 8 + w;
  bool valid = e < E;
  int s = 0, t = 0;
  if (valid) { s = src[e]; t = tgt[e]; }
  sh[w][lane]      = valid ? h[(size_t)s * 64 + lane]      : 0.f;
  sh[w][lane + 32] = valid ? h[(size_t)s * 64 + lane + 32] : 0.f;
  __syncthreads();
  if (!valid) return;  // wave-uniform
  const _Float16* ep = ew + (size_t)e * 4096;
  __builtin_prefetch(ep + 2048, 0, 0);  // global_prefetch of 2nd half of the tile
  const int rsel = lane >> 4;        // 0: even d rows, 1: odd d rows
  const int o4 = (lane & 15) * 4;    // 4 output columns per lane
  float acc0 = 0.f, acc1 = 0.f, acc2 = 0.f, acc3 = 0.f;
#pragma unroll 8
  for (int dd = 0; dd < 64; dd += 2) {
    int d = dd + rsel;
    float v = sh[w][d];
    h4 hv = __builtin_nontemporal_load((const h4*)(ep + d * 64 + o4));
    acc0 += v * (float)hv.x;
    acc1 += v * (float)hv.y;
    acc2 += v * (float)hv.z;
    acc3 += v * (float)hv.w;
  }
  acc0 += __shfl_xor(acc0, 16);
  acc1 += __shfl_xor(acc1, 16);
  acc2 += __shfl_xor(acc2, 16);
  acc3 += __shfl_xor(acc3, 16);
  if (rsel == 0) {
    float* ap = agg + (size_t)t * 64 + o4;
    atomicAdd(ap + 0, acc0);
    atomicAdd(ap + 1, acc1);
    atomicAdd(ap + 2, acc2);
    atomicAdd(ap + 3, acc3);
  }
}

// ---- fused node update: conv + leaky + GRU cell, WMMA for all matmuls -------
#define GSTR 193
__global__ void __launch_bounds__(128)
node_update_kernel(float* __restrict__ h, const float* __restrict__ agg,
                   const float* __restrict__ deg, const _Float16* __restrict__ rootpk,
                   const _Float16* __restrict__ wihpk, const _Float16* __restrict__ whhpk,
                   const float* __restrict__ conv_b, const float* __restrict__ b_ih,
                   const float* __restrict__ b_hh, int N) {
  __shared__ float sh[16 * 65];
  __shared__ float sm[16 * 65];
  __shared__ float sgi[16 * GSTR];
  __shared__ float sgh[16 * GSTR];
  int n0 = blockIdx.x * 16;
  int tid = threadIdx.x;
  for (int i = tid; i < 16 * 64; i += 128) {
    int r = i >> 6, c = i & 63;
    sh[r * 65 + c] = (n0 + r < N) ? h[(size_t)(n0 + r) * 64 + c] : 0.f;
  }
  __syncthreads();
  int lane = tid & 31, w = tid >> 5;
  int m = lane & 15;
  int mb = (lane >> 4) << 3;
  v16h ah0, ah1;
#pragma unroll
  for (int e = 0; e < 16; e++) {
    ah0[e] = (_Float16)sh[m * 65 + kidx(lane, e)];
    ah1[e] = (_Float16)sh[m * 65 + 32 + kidx(lane, e)];
  }
  // conv = agg*inv_deg + h @ root_w + conv_b ; m = leaky(conv). 4 waves, 4 N-tiles
  {
    int t = w;
    v16h b0 = *(const v16h*)(rootpk + ((size_t)t * 32 + lane) * 16);
    v16h b1 = *(const v16h*)(rootpk + ((size_t)(4 + t) * 32 + lane) * 16);
    v8f c = {};
    c = wmma32(ah0, b0, c);
    c = wmma32(ah1, b1, c);
    int n = t * 16 + (lane & 15);
    float cb = conv_b[n];
#pragma unroll
    for (int r = 0; r < 8; r++) {
      int row = mb + r, node = n0 + row;
      float aggv = 0.f, inv = 1.f;
      if (node < N) {
        inv = 1.f / fmaxf(deg[node], 1.f);
        aggv = agg[(size_t)node * 64 + n];
      }
      sm[row * 65 + n] = leaky(c[r] + aggv * inv + cb);
    }
  }
  __syncthreads();
  v16h am0, am1;
#pragma unroll
  for (int e = 0; e < 16; e++) {
    am0[e] = (_Float16)sm[m * 65 + kidx(lane, e)];
    am1[e] = (_Float16)sm[m * 65 + 32 + kidx(lane, e)];
  }
  // gi = m @ w_ih^T + b_ih (12 tiles of 16 cols)
  for (int t = w; t < 12; t += 4) {
    v16h b0 = *(const v16h*)(wihpk + ((size_t)t * 32 + lane) * 16);
    v16h b1 = *(const v16h*)(wihpk + ((size_t)(12 + t) * 32 + lane) * 16);
    v8f c = {};
    c = wmma32(am0, b0, c);
    c = wmma32(am1, b1, c);
    int n = t * 16 + (lane & 15);
    float bb = b_ih[n];
#pragma unroll
    for (int r = 0; r < 8; r++) sgi[(mb + r) * GSTR + n] = c[r] + bb;
  }
  // gh = h @ w_hh^T + b_hh
  for (int t = w; t < 12; t += 4) {
    v16h b0 = *(const v16h*)(whhpk + ((size_t)t * 32 + lane) * 16);
    v16h b1 = *(const v16h*)(whhpk + ((size_t)(12 + t) * 32 + lane) * 16);
    v8f c = {};
    c = wmma32(ah0, b0, c);
    c = wmma32(ah1, b1, c);
    int n = t * 16 + (lane & 15);
    float bb = b_hh[n];
#pragma unroll
    for (int r = 0; r < 8; r++) sgh[(mb + r) * GSTR + n] = c[r] + bb;
  }
  __syncthreads();
  // GRU gates + state update (in place: this block owns its 16 rows)
  for (int i = tid; i < 16 * 64; i += 128) {
    int r = i >> 6, j = i & 63;
    int node = n0 + r;
    if (node >= N) continue;
    float gir = sgi[r * GSTR + j], giz = sgi[r * GSTR + 64 + j], gin = sgi[r * GSTR + 128 + j];
    float ghr = sgh[r * GSTR + j], ghz = sgh[r * GSTR + 64 + j], ghn = sgh[r * GSTR + 128 + j];
    float rg = sigm(gir + ghr);
    float zg = sigm(giz + ghz);
    float ng = tanhf(gin + rg * ghn);
    float hp = sh[r * 65 + j];
    h[(size_t)node * 64 + j] = (1.f - zg) * ng + zg * hp;
  }
}

// ---- Set2Set: q vector from LSTM biases (inputs are all-zero) ---------------
__global__ void qvec_kernel(const float* __restrict__ bih, const float* __restrict__ bhh,
                            float* __restrict__ q) {
  int j = threadIdx.x;
  if (j >= 64) return;
  float gi = bih[j] + bhh[j];
  float gg = bih[128 + j] + bhh[128 + j];
  float go = bih[192 + j] + bhh[192 + j];
  float c = sigm(gi) * tanhf(gg);
  q[j] = sigm(go) * tanhf(c);
}

__global__ void attn_dot_kernel(const float* __restrict__ h, const float* __restrict__ q,
                                const int* __restrict__ batch, float* __restrict__ evec,
                                float* __restrict__ emax, int N) {
  __shared__ float sq[64];
  if (threadIdx.x < 64) sq[threadIdx.x] = q[threadIdx.x];
  __syncthreads();
  int n = blockIdx.x * blockDim.x + threadIdx.x;
  if (n >= N) return;
  float acc = 0.f;
#pragma unroll
  for (int j = 0; j < 64; j++) acc += h[(size_t)n * 64 + j] * sq[j];
  evec[n] = acc;
  atomicMaxF(&emax[batch[n]], acc);
}

__global__ void emax_fix_kernel(float* __restrict__ emax, int B) {
  int b = blockIdx.x * blockDim.x + threadIdx.x;
  if (b >= B) return;
  float v = emax[b];
  if (!(fabsf(v) < INFINITY)) emax[b] = 0.f;
}

__global__ void attn_exp_kernel(const float* __restrict__ evec, const float* __restrict__ emax,
                                const int* __restrict__ batch, float* __restrict__ ex,
                                float* __restrict__ denom, int N) {
  int n = blockIdx.x * blockDim.x + threadIdx.x;
  if (n >= N) return;
  int b = batch[n];
  float v = expf(evec[n] - emax[b]);
  ex[n] = v;
  atomicAdd(&denom[b], v);
}

__global__ void attn_read_kernel(const float* __restrict__ h, const float* __restrict__ ex,
                                 const float* __restrict__ denom, const int* __restrict__ batch,
                                 float* __restrict__ rread, int N) {
  int idx = blockIdx.x * blockDim.x + threadIdx.x;
  if (idx >= N * DIM) return;
  int n = idx >> 6, j = idx & 63;
  int b = batch[n];
  float wgt = ex[n] / fmaxf(denom[b], 1e-16f);
  atomicAdd(&rread[(size_t)b * 64 + j], wgt * h[idx]);
}

__global__ void out_kernel(const float* __restrict__ q, const float* __restrict__ rread,
                           const float* __restrict__ lw, const float* __restrict__ lb,
                           float* __restrict__ out) {
  int idx = blockIdx.x * blockDim.x + threadIdx.x;
  if (idx >= NUM_GRAPHS * OUT_DIM) return;
  int b = idx >> 4, o = idx & 15;
  float acc = lb[o];
#pragma unroll
  for (int j = 0; j < 64; j++) acc += q[j] * lw[o * 128 + j];
#pragma unroll
  for (int j = 0; j < 64; j++) acc += rread[(size_t)b * 64 + j] * lw[o * 128 + 64 + j];
  out[idx] = acc;
}

extern "C" void kernel_launch(void* const* d_in, const int* in_sizes, int n_in,
                              void* d_out, int out_size, void* d_ws, size_t ws_size,
                              hipStream_t stream) {
  const float* x         = (const float*)d_in[0];
  const float* edge_attr = (const float*)d_in[1];
  const int*   eidx      = (const int*)d_in[2];
  const int*   batch     = (const int*)d_in[3];
  const float* lin0_w    = (const float*)d_in[4];
  const float* lin0_b    = (const float*)d_in[5];
  const float* net_w1    = (const float*)d_in[6];
  const float* net_b1    = (const float*)d_in[7];
  const float* net_w2    = (const float*)d_in[8];
  const float* net_b2    = (const float*)d_in[9];
  const float* root_w    = (const float*)d_in[10];
  const float* conv_b    = (const float*)d_in[11];
  const float* gru_w_ih  = (const float*)d_in[12];
  const float* gru_w_hh  = (const float*)d_in[13];
  const float* gru_b_ih  = (const float*)d_in[14];
  const float* gru_b_hh  = (const float*)d_in[15];
  const float* lstm_b_ih = (const float*)d_in[18];
  const float* lstm_b_hh = (const float*)d_in[19];
  const float* lout_w    = (const float*)d_in[20];
  const float* lout_b    = (const float*)d_in[21];

  const int N = in_sizes[0] / NUM_FEAT;
  const int E = in_sizes[1] / EDGE_FEAT;
  const int* src = eidx;
  const int* tgt = eidx + E;

  char* ws = (char*)d_ws;
  size_t off = 0;
  auto carve = [&](size_t bytes) -> void* {
    void* p = (void*)(ws + off);
    off = (off + bytes + 255) & ~(size_t)255;
    return p;
  };
  _Float16* ewbuf  = (_Float16*)carve((size_t)E * 4096 * sizeof(_Float16));
  float*    hidden = (float*)carve((size_t)E * 64 * sizeof(float));
  _Float16* w2pk   = (_Float16*)carve((size_t)2 * 256 * 512 * sizeof(_Float16));
  _Float16* rootpk = (_Float16*)carve((size_t)2 * 4 * 512 * sizeof(_Float16));
  _Float16* wihpk  = (_Float16*)carve((size_t)2 * 12 * 512 * sizeof(_Float16));
  _Float16* whhpk  = (_Float16*)carve((size_t)2 * 12 * 512 * sizeof(_Float16));
  float*    h      = (float*)carve((size_t)N * 64 * sizeof(float));
  float*    agg    = (float*)carve((size_t)N * 64 * sizeof(float));
  float*    deg    = (float*)carve((size_t)N * sizeof(float));
  float*    qv     = (float*)carve(64 * sizeof(float));
  float*    evec   = (float*)carve((size_t)N * sizeof(float));
  float*    exv    = (float*)carve((size_t)N * sizeof(float));
  float*    emax   = (float*)carve(NUM_GRAPHS * sizeof(float));
  float*    denom  = (float*)carve(NUM_GRAPHS * sizeof(float));
  float*    rread  = (float*)carve((size_t)NUM_GRAPHS * 64 * sizeof(float));
  (void)ws_size; (void)n_in; (void)out_size;

  // weight packs (once)
  pack_b_kernel<<<(2 * 256 * 512 + 255) / 256, 256, 0, stream>>>(net_w2, w2pk, 256, 1, 64);
  pack_b_kernel<<<(2 * 4 * 512 + 255) / 256, 256, 0, stream>>>(root_w, rootpk, 4, 64, 1);
  pack_b_kernel<<<(2 * 12 * 512 + 255) / 256, 256, 0, stream>>>(gru_w_ih, wihpk, 12, 1, 64);
  pack_b_kernel<<<(2 * 12 * 512 + 255) / 256, 256, 0, stream>>>(gru_w_hh, whhpk, 12, 1, 64);

  // edge network (once): hidden, then the big WMMA GEMM producing ew in f16
  edge_hidden_kernel<<<(E * 64 + 255) / 256, 256, 0, stream>>>(edge_attr, net_w1, net_b1, hidden, E);
  ew_gemm_kernel<<<(E + 15) / 16, 256, 0, stream>>>(hidden, w2pk, net_b2, ewbuf, E);

  // node init + degrees
  lin0_kernel<<<(N * 64 + 255) / 256, 256, 0, stream>>>(x, lin0_w, lin0_b, h, N);
  fill_kernel<<<(N + 255) / 256, 256, 0, stream>>>(deg, 0.f, (size_t)N);
  deg_kernel<<<(E + 255) / 256, 256, 0, stream>>>(tgt, deg, E);

  // 6 message-passing + GRU steps
  for (int it = 0; it < 6; ++it) {
    fill_kernel<<<(unsigned)(((size_t)N * 64 + 255) / 256), 256, 0, stream>>>(agg, 0.f, (size_t)N * 64);
    msg_scatter_kernel<<<(E + 7) / 8, 256, 0, stream>>>(h, ewbuf, src, tgt, agg, E);
    node_update_kernel<<<(N + 15) / 16, 128, 0, stream>>>(h, agg, deg, rootpk, wihpk, whhpk,
                                                          conv_b, gru_b_ih, gru_b_hh, N);
  }

  // Set2Set (1 processing step) + readout
  qvec_kernel<<<1, 64, 0, stream>>>(lstm_b_ih, lstm_b_hh, qv);
  fill_kernel<<<(NUM_GRAPHS + 255) / 256, 256, 0, stream>>>(emax, -INFINITY, (size_t)NUM_GRAPHS);
  fill_kernel<<<(NUM_GRAPHS + 255) / 256, 256, 0, stream>>>(denom, 0.f, (size_t)NUM_GRAPHS);
  fill_kernel<<<(NUM_GRAPHS * 64 + 255) / 256, 256, 0, stream>>>(rread, 0.f, (size_t)NUM_GRAPHS * 64);
  attn_dot_kernel<<<(N + 255) / 256, 256, 0, stream>>>(h, qv, batch, evec, emax, N);
  emax_fix_kernel<<<(NUM_GRAPHS + 63) / 64, 64, 0, stream>>>(emax, NUM_GRAPHS);
  attn_exp_kernel<<<(N + 255) / 256, 256, 0, stream>>>(evec, emax, batch, exv, denom, N);
  attn_read_kernel<<<(N * 64 + 255) / 256, 256, 0, stream>>>(h, exv, denom, batch, rread, N);
  out_kernel<<<(NUM_GRAPHS * OUT_DIM + 255) / 256, 256, 0, stream>>>(qv, rread, lout_w, lout_b,
                                                                     (float*)d_out);
}
